// MultiHeadAttention_56392920596740
// MI455X (gfx1250) — matrix operand order repaired and checked by reading, not compiled, not run on hardware
//
#include <hip/hip_runtime.h>
#include <hip/hip_bf16.h>

typedef __bf16 v16bf __attribute__((ext_vector_type(16)));
typedef __bf16 bfx8  __attribute__((ext_vector_type(8)));
typedef float  v8f   __attribute__((ext_vector_type(8)));
typedef unsigned int u32;
typedef int v4i_vs __attribute__((vector_size(4 * sizeof(int))));

#define B_  4
#define S_  2048
#define D_  1024
#define H_  16
#define HD_ 64
#define KC_ 64   // K chunk staged per LDS buffer (two 32-deep WMMA steps)

// ---------------- Async global->LDS (CDNA5), guarded ------------------------
#if defined(__HIP_DEVICE_COMPILE__) && \
    __has_builtin(__builtin_amdgcn_global_load_async_to_lds_b128)
#define USE_ASYNC_LDS 1
#else
#define USE_ASYNC_LDS 0
#endif

#if USE_ASYNC_LDS
typedef __attribute__((address_space(1))) v4i_vs* gvec_ptr;
typedef __attribute__((address_space(3))) v4i_vs* lvec_ptr;
#endif

__device__ __forceinline__ void async_copy16(void* lds, const void* gmem) {
#if USE_ASYNC_LDS
  __builtin_amdgcn_global_load_async_to_lds_b128((gvec_ptr)gmem, (lvec_ptr)lds,
                                                 0, 0);
#else
  *(uint4*)lds = *(const uint4*)gmem;
#endif
}

__device__ __forceinline__ void wait_async() {
#if USE_ASYNC_LDS
#if __has_builtin(__builtin_amdgcn_s_wait_asynccnt)
  __builtin_amdgcn_s_wait_asynccnt(0);
#else
  asm volatile("s_wait_asynccnt 0x0" ::: "memory");
#endif
#endif
}

// ---------------- WMMA helpers (CDNA5 16x16x32 bf16, f32 accum) -------------

__device__ __forceinline__ v8f wmma_bf16(v16bf a, v16bf b, v8f c) {
  return __builtin_amdgcn_wmma_f32_16x16x32_bf16(false, a, false, b, (short)0, c,
                                                 false, false);
}

__device__ __forceinline__ v16bf concat8(bfx8 lo, bfx8 hi) {
  return __builtin_shufflevector(lo, hi, 0, 1, 2, 3, 4, 5, 6, 7, 8, 9, 10, 11,
                                 12, 13, 14, 15);
}

// A-matrix 16x32 (MxK) from LDS row-major [m][k]; both per-lane K runs are
// contiguous -> two ds_load_b128. ld must be a multiple of 8 elements.
__device__ __forceinline__ v16bf load_A16x32(const __bf16* lds, int ld, int lane) {
  const int m  = lane & 15;
  const int hi = (lane >> 4) << 3;
  const __bf16* p = lds + m * ld + hi;
  return concat8(*(const bfx8*)p, *(const bfx8*)(p + 16));
}

// B-matrix 32x16 (KxN) from a TRANSPOSED LDS tile stored [n][k] row-major.
__device__ __forceinline__ v16bf load_B32x16(const __bf16* lds, int ld, int lane,
                                             int n0, int k0) {
  const int n  = (lane & 15) + n0;
  const int kb = ((lane >> 4) << 3) + k0;
  const __bf16* p = lds + n * ld + kb;
  return concat8(*(const bfx8*)p, *(const bfx8*)(p + 16));
}

__device__ __forceinline__ u32 pack2bf(float a, float b) {
  union { u32 u; __bf16 h[2]; } cv;
  cv.h[0] = (__bf16)a; cv.h[1] = (__bf16)b;
  return cv.u;
}

__device__ __forceinline__ void split2bf(u32 w, __bf16& a, __bf16& b) {
  union { u32 u; __bf16 h[2]; } cv;
  cv.u = w;
  a = cv.h[0]; b = cv.h[1];
}

// ---------------- Kernel 1: QKV projections -> [B,H,S,HD] bf16 --------------
// WG tile 128x128, K staged in 64-chunks, double buffered. 4 waves; wave w
// owns rows 32w..32w+31 (mt=0,1) x 128 cols (nt=0..7): 32 WMMA per chunk.
__global__ __launch_bounds__(128) void qkv_gemm_kernel(
    const float* __restrict__ x, const float* __restrict__ Wq,
    const float* __restrict__ Wk, const float* __restrict__ Wv,
    __bf16* __restrict__ qkv) {
  const int n1 = blockIdx.x * 128;
  const int m0 = blockIdx.y * 128;
  const int which = blockIdx.z;
  const float* W = (which == 0) ? Wq : (which == 1) ? Wk : Wv;
  __bf16* out = qkv + (size_t)which * (size_t)B_ * S_ * D_;

  __shared__ __align__(16) __bf16 As[2][128][72];  // [m][k]
  __shared__ __align__(16) __bf16 Bs[2][128][72];  // [n][k] (transposed W)

  const int t = threadIdx.x;
  const int wave = t >> 5, lane = t & 31;
  v8f acc[2][8] = {};

  auto stage = [&](int buf, int k0) {
#pragma unroll
    for (int i = 0; i < 16; ++i) {  // A: 128x64 fp32, float4 loads
      const int idx = t + 128 * i;
      const int r = idx >> 4, c4 = idx & 15;
      const float4 f = ((const float4*)(x + (size_t)(m0 + r) * D_ + k0))[c4];
      uint2 pk; pk.x = pack2bf(f.x, f.y); pk.y = pack2bf(f.z, f.w);
      *((uint2*)&As[buf][r][0] + c4) = pk;
    }
#pragma unroll
    for (int i = 0; i < 16; ++i) {  // B: 64x128 fp32 -> [n][k]
      const int idx = t + 128 * i;
      const int n = idx & 127, g = idx >> 7;  // g: k-quad 0..15
      const float* wp = W + (size_t)(k0 + 4 * g) * D_ + n1 + n;
      uint2 pk; pk.x = pack2bf(wp[0], wp[D_]);
      pk.y = pack2bf(wp[2 * (size_t)D_], wp[3 * (size_t)D_]);
      *((uint2*)&Bs[buf][n][0] + g) = pk;
    }
  };

  auto compute = [&](int buf) {
#pragma unroll
    for (int ks = 0; ks < 2; ++ks) {
      v16bf a[2];
#pragma unroll
      for (int mt = 0; mt < 2; ++mt)
        a[mt] = load_A16x32(&As[buf][wave * 32 + mt * 16][ks * 32], 72, lane);
      // software-pipelined B fragments: load nt+1 before consuming nt
      v16bf bcur = load_B32x16(&Bs[buf][0][0], 72, lane, 0, ks * 32);
#pragma unroll
      for (int nt = 0; nt < 8; ++nt) {
        v16bf bnext = bcur;
        if (nt < 7)
          bnext = load_B32x16(&Bs[buf][0][0], 72, lane, (nt + 1) * 16, ks * 32);
#pragma unroll
        for (int mt = 0; mt < 2; ++mt)
          acc[mt][nt] = wmma_bf16(a[mt], bcur, acc[mt][nt]);
        bcur = bnext;
      }
    }
  };

  stage(0, 0);
  __syncthreads();
  for (int kc = 0; kc < D_ / KC_; ++kc) {
    if (kc + 1 < D_ / KC_) stage((kc + 1) & 1, (kc + 1) * KC_);
    compute(kc & 1);
    __syncthreads();
  }

  // store permuted to [B,H,S,HD]
  const int hi8 = (lane >> 4) << 3;
  const int l15 = lane & 15;
#pragma unroll
  for (int mt = 0; mt < 2; ++mt)
#pragma unroll
    for (int nt = 0; nt < 8; ++nt) {
      const int n = n1 + nt * 16 + l15;
      const int h = n >> 6, hd = n & 63;
#pragma unroll
      for (int i = 0; i < 8; ++i) {
        const int m = m0 + wave * 32 + mt * 16 + hi8 + i;
        const int bb = m >> 11, s = m & (S_ - 1);
        out[(((size_t)bb * H_ + h) * S_ + s) * HD_ + hd] = (__bf16)acc[mt][nt][i];
      }
    }
}

// ---------------- Kernel 2: causal flash attention --------------------------
// grid (S/128, B*H), block 128. Wave w owns queries q0+32w..q0+32w+31
// (query tiles qt=0,1). 32-key blocks, double-buffered K/V staging.
__global__ __launch_bounds__(128) void flash_attn_kernel(
    const __bf16* __restrict__ Qg, const __bf16* __restrict__ Kg,
    const __bf16* __restrict__ Vg, __bf16* __restrict__ ctx) {
  const int qb = blockIdx.x;
  const int bh = blockIdx.y;
  const int b = bh >> 4, h = bh & 15;
  const int q0 = qb * 128;

  const __bf16* Qp = Qg + (size_t)bh * S_ * HD_;
  const __bf16* Kp = Kg + (size_t)bh * S_ * HD_;
  const __bf16* Vp = Vg + (size_t)bh * S_ * HD_;

  __shared__ __align__(16) __bf16 Qs[128][72];     // [query][hd]
  __shared__ __align__(16) __bf16 Ks[2][32][72];   // [key][hd]  (B^T for QK^T)
  __shared__ __align__(16) __bf16 Vt[2][64][40];   // [hd][key]  (B^T for PV)
  __shared__ __align__(16) __bf16 Ps[4][32][40];   // per-wave P [row][key]

  const int t = threadIdx.x;
  const int wave = t >> 5, lane = t & 31;
  const int hi8 = (lane >> 4) << 3;
  const int l15 = lane & 15;

  {  // stage Q block: 128x64 bf16 = 1024 x 16B (async when available)
    const uint4* Qu = (const uint4*)(Qp + (size_t)q0 * HD_);
#pragma unroll
    for (int i = 0; i < 8; ++i) {
      const int idx = t + 128 * i;
      const int r = idx >> 3, c4 = idx & 7;
      async_copy16(&Qs[r][8 * c4], &Qu[r * 8 + c4]);
    }
  }

  auto stageKV = [&](int buf, int kbeg) {
    const uint4* Ku = (const uint4*)(Kp + (size_t)kbeg * HD_);
#pragma unroll
    for (int i = 0; i < 2; ++i) {  // K: 32x64 bf16 = 256 x 16B (async)
      const int idx = t + 128 * i;
      const int r = idx >> 3, c4 = idx & 7;
      async_copy16(&Ks[buf][r][8 * c4], &Ku[r * 8 + c4]);
    }
    const u32* Vu = (const u32*)(Vp + (size_t)kbeg * HD_);
#pragma unroll
    for (int i = 0; i < 8; ++i) {  // V: transpose scatter into [hd][key]
      const int idx = t + 128 * i;
      const int r = idx >> 5, c = idx & 31;  // r: key, c: hd-pair
      __bf16 v0, v1;
      split2bf(Vu[r * 32 + c], v0, v1);
      Vt[buf][2 * c][r] = v0;
      Vt[buf][2 * c + 1][r] = v1;
    }
  };

  stageKV(0, 0);
  wait_async();
  __syncthreads();

  v16bf qa[2][2];  // [query tile][k-step]
#pragma unroll
  for (int qt = 0; qt < 2; ++qt)
#pragma unroll
    for (int ks = 0; ks < 2; ++ks)
      qa[qt][ks] = load_A16x32(&Qs[wave * 32 + qt * 16][ks * 32], 72, lane);

  float mrow[2][8], lrow[2][8];
  v8f O[2][4] = {};
#pragma unroll
  for (int qt = 0; qt < 2; ++qt)
#pragma unroll
    for (int i = 0; i < 8; ++i) { mrow[qt][i] = -1e30f; lrow[qt][i] = 0.f; }

  // softmax in exp2 domain: scale = log2(e)/sqrt(HD)
  const float scale2 = 0.125f * 1.44269504088896f;

  const int jmax = (q0 + 127) >> 5;
  for (int kb = 0; kb <= jmax; ++kb) {
    const int kbeg = kb * 32;
    const int buf = kb & 1;
    if (kb < jmax) stageKV(buf ^ 1, kbeg + 32);

    // scores: S[qt][kt] (16q x 16k), HD contraction in two 32-steps.
    // Pipelined over the 4 (ks,kt) fragments.
    v8f sc[2][2] = {};
    v16bf bk = load_B32x16(&Ks[buf][0][0], 72, lane, 0, 0);
#pragma unroll
    for (int p = 0; p < 4; ++p) {
      const int ks = p >> 1, kt = p & 1;
      v16bf bnext = bk;
      if (p < 3) {
        const int pn = p + 1;
        bnext = load_B32x16(&Ks[buf][0][0], 72, lane, (pn & 1) * 16,
                            (pn >> 1) * 32);
      }
#pragma unroll
      for (int qt = 0; qt < 2; ++qt)
        sc[qt][kt] = wmma_bf16(qa[qt][ks], bk, sc[qt][kt]);
      bk = bnext;
    }

#pragma unroll
    for (int qt = 0; qt < 2; ++qt) {
      float sv0[8], sv1[8], rmax[8];
#pragma unroll
      for (int i = 0; i < 8; ++i) {
        const int q = q0 + wave * 32 + qt * 16 + hi8 + i;
        sv0[i] = (kbeg + l15 <= q) ? sc[qt][0][i] * scale2 : -1e30f;
        sv1[i] = (kbeg + 16 + l15 <= q) ? sc[qt][1][i] * scale2 : -1e30f;
        rmax[i] = fmaxf(sv0[i], sv1[i]);
      }
#pragma unroll
      for (int msk = 1; msk <= 8; msk <<= 1)
#pragma unroll
        for (int i = 0; i < 8; ++i)
          rmax[i] = fmaxf(rmax[i], __shfl_xor(rmax[i], msk, 32));

      float corr[8], rs[8];
#pragma unroll
      for (int i = 0; i < 8; ++i) {
        const float mn = fmaxf(mrow[qt][i], rmax[i]);
        corr[i] = exp2f(mrow[qt][i] - mn);
        mrow[qt][i] = mn;
        const float p0 = exp2f(sv0[i] - mn);
        const float p1 = exp2f(sv1[i] - mn);
        Ps[wave][qt * 16 + hi8 + i][l15] = (__bf16)p0;
        Ps[wave][qt * 16 + hi8 + i][16 + l15] = (__bf16)p1;
        rs[i] = p0 + p1;
      }
#pragma unroll
      for (int msk = 1; msk <= 8; msk <<= 1)
#pragma unroll
        for (int i = 0; i < 8; ++i) rs[i] += __shfl_xor(rs[i], msk, 32);
#pragma unroll
      for (int i = 0; i < 8; ++i) lrow[qt][i] = lrow[qt][i] * corr[i] + rs[i];
#pragma unroll
      for (int nt = 0; nt < 4; ++nt)
#pragma unroll
        for (int i = 0; i < 8; ++i) O[qt][nt][i] *= corr[i];
    }

    // O += P (16x32) x V (32x64); V fragments shared across query tiles.
    v16bf pa[2];
#pragma unroll
    for (int qt = 0; qt < 2; ++qt)
      pa[qt] = load_A16x32(&Ps[wave][qt * 16][0], 40, lane);
    v16bf vb = load_B32x16(&Vt[buf][0][0], 40, lane, 0, 0);
#pragma unroll
    for (int nt = 0; nt < 4; ++nt) {
      v16bf vnext = vb;
      if (nt < 3)
        vnext = load_B32x16(&Vt[buf][0][0], 40, lane, (nt + 1) * 16, 0);
#pragma unroll
      for (int qt = 0; qt < 2; ++qt)
        O[qt][nt] = wmma_bf16(pa[qt], vb, O[qt][nt]);
      vb = vnext;
    }

    wait_async();
    __syncthreads();
  }

  // normalize and write ctx as [B,S,D] bf16 (D col = h*64 + hd)
#pragma unroll
  for (int qt = 0; qt < 2; ++qt)
#pragma unroll
    for (int i = 0; i < 8; ++i) {
      const float inv = 1.0f / lrow[qt][i];
      const int q = q0 + wave * 32 + qt * 16 + hi8 + i;
#pragma unroll
      for (int nt = 0; nt < 4; ++nt) {
        const int hd = nt * 16 + l15;
        ctx[((size_t)b * S_ + q) * D_ + h * HD_ + hd] =
            (__bf16)(O[qt][nt][i] * inv);
      }
    }
}

// ---------------- Kernel 3: output projection + bias ------------------------
// WG tile 128x128, K=64 chunks, double buffered. grid (D/128, B*S/128).
__global__ __launch_bounds__(128) void out_gemm_kernel(
    const __bf16* __restrict__ Ag, const float* __restrict__ Wo,
    const float* __restrict__ bo, float* __restrict__ out) {
  const int n1 = blockIdx.x * 128;
  const int m0 = blockIdx.y * 128;

  __shared__ __align__(16) __bf16 As[2][128][72];  // [m][k]
  __shared__ __align__(16) __bf16 Bs[2][128][72];  // [n][k]

  const int t = threadIdx.x;
  const int wave = t >> 5, lane = t & 31;
  v8f acc[2][8] = {};

  auto stage = [&](int buf, int k0) {
#pragma unroll
    for (int i = 0; i < 8; ++i) {  // A: 128x64 bf16 = 1024 x 16B (async)
      const int idx = t + 128 * i;
      const int r = idx >> 3, c4 = idx & 7;
      async_copy16(&As[buf][r][8 * c4],
                   ((const uint4*)(Ag + (size_t)(m0 + r) * D_)) + (k0 >> 3) + c4);
    }
#pragma unroll
    for (int i = 0; i < 16; ++i) {  // B: 64x128 fp32 -> [n][k]
      const int idx = t + 128 * i;
      const int n = idx & 127, g = idx >> 7;
      const float* wp = Wo + (size_t)(k0 + 4 * g) * D_ + n1 + n;
      uint2 pk; pk.x = pack2bf(wp[0], wp[D_]);
      pk.y = pack2bf(wp[2 * (size_t)D_], wp[3 * (size_t)D_]);
      *((uint2*)&Bs[buf][n][0] + g) = pk;
    }
  };

  auto compute = [&](int buf) {
#pragma unroll
    for (int ks = 0; ks < 2; ++ks) {
      v16bf a[2];
#pragma unroll
      for (int mt = 0; mt < 2; ++mt)
        a[mt] = load_A16x32(&As[buf][wave * 32 + mt * 16][ks * 32], 72, lane);
      v16bf bcur = load_B32x16(&Bs[buf][0][0], 72, lane, 0, ks * 32);
#pragma unroll
      for (int nt = 0; nt < 8; ++nt) {
        v16bf bnext = bcur;
        if (nt < 7)
          bnext = load_B32x16(&Bs[buf][0][0], 72, lane, (nt + 1) * 16, ks * 32);
#pragma unroll
        for (int mt = 0; mt < 2; ++mt)
          acc[mt][nt] = wmma_bf16(a[mt], bcur, acc[mt][nt]);
        bcur = bnext;
      }
    }
  };

  stage(0, 0);
  wait_async();
  __syncthreads();
  for (int kc = 0; kc < D_ / KC_; ++kc) {
    if (kc + 1 < D_ / KC_) stage((kc + 1) & 1, (kc + 1) * KC_);
    compute(kc & 1);
    wait_async();
    __syncthreads();
  }

  const int hi8 = (lane >> 4) << 3;
  const int l15 = lane & 15;
#pragma unroll
  for (int mt = 0; mt < 2; ++mt)
#pragma unroll
    for (int nt = 0; nt < 8; ++nt) {
      const int n = n1 + nt * 16 + l15;
      const float bias = bo[n];
#pragma unroll
      for (int i = 0; i < 8; ++i) {
        const int m = m0 + wave * 32 + mt * 16 + hi8 + i;
        out[(size_t)m * D_ + n] = acc[mt][nt][i] + bias;
      }
    }
}

// ---------------- Launch ----------------------------------------------------
extern "C" void kernel_launch(void* const* d_in, const int* in_sizes, int n_in,
                              void* d_out, int out_size, void* d_ws, size_t ws_size,
                              hipStream_t stream) {
  const float* x  = (const float*)d_in[0];
  const float* Wq = (const float*)d_in[1];
  const float* Wk = (const float*)d_in[2];
  const float* Wv = (const float*)d_in[3];
  const float* Wo = (const float*)d_in[4];
  const float* bo = (const float*)d_in[5];
  float* out = (float*)d_out;

  const size_t BSD = (size_t)B_ * S_ * D_;  // 8,388,608 elems
  __bf16* ws  = (__bf16*)d_ws;
  __bf16* Qb  = ws;            // [B,H,S,HD]
  __bf16* Kb  = ws + BSD;      // [B,H,S,HD]
  __bf16* Vb  = ws + 2 * BSD;  // [B,H,S,HD]
  __bf16* ctx = ws + 3 * BSD;  // [B,S,D]

  dim3 gProj(D_ / 128, (B_ * S_) / 128, 3);
  qkv_gemm_kernel<<<gProj, 128, 0, stream>>>(x, Wq, Wk, Wv, Qb);

  dim3 gAttn(S_ / 128, B_ * H_);
  flash_attn_kernel<<<gAttn, 128, 0, stream>>>(Qb, Kb, Vb, ctx);

  dim3 gOut(D_ / 128, (B_ * S_) / 128);
  out_gemm_kernel<<<gOut, 128, 0, stream>>>(ctx, Wo, bo, out);
}
